// PersonaEmotionXModel_10024453669599
// MI455X (gfx1250) — matrix operand order repaired and checked by reading, not compiled, not run on hardware
//
#include <hip/hip_runtime.h>
#include <stdint.h>

// ---------------- types for CDNA5 WMMA ----------------
typedef __bf16 v16bf __attribute__((ext_vector_type(16)));
typedef __bf16 v8bf  __attribute__((ext_vector_type(8)));
typedef float  v8f   __attribute__((ext_vector_type(8)));

union AFrag { v16bf v; struct Halves { v8bf lo, hi; } p; };

// ---------------- problem constants ----------------
constexpr int kS    = 65536;
constexpr int kN    = 8192;
constexpr int kSP   = 512;
constexpr int kD    = 768;
constexpr int kF    = 1536;  // 2*D
constexpr int kH1   = 384;
constexpr int kOut  = 5;

__device__ __forceinline__ unsigned short f2bf(float f) {
    unsigned int u = __builtin_bit_cast(unsigned int, f);
    u += 0x7FFFu + ((u >> 16) & 1u);          // round-to-nearest-even
    return (unsigned short)(u >> 16);
}

// ---------------------------------------------------------------
// Convert fp32 -> bf16, optionally also writing a transposed copy.
// src: [R][C] row-major fp32.  dst_rm: [R][C] bf16.  dst_tr: [C][R] bf16.
// ---------------------------------------------------------------
__global__ void conv_kernel(const float* __restrict__ src,
                            unsigned short* __restrict__ dst_rm,
                            unsigned short* __restrict__ dst_tr,
                            int R, int C) {
    int i = blockIdx.x * blockDim.x + threadIdx.x;
    if (i >= R * C) return;
    int r = i / C, c = i - r * C;
    unsigned short b = f2bf(src[i]);
    if (dst_rm) dst_rm[i] = b;
    if (dst_tr) dst_tr[(size_t)c * R + r] = b;
}

// ---------------------------------------------------------------
// Segment-mean pooling. starts are fixed at 8*seg; len = end-start in [1,8].
// Writes bf16 pooled into final_bf[:, 0:768].
// ---------------------------------------------------------------
__global__ void pool_kernel(const float* __restrict__ emb,
                            const int* __restrict__ sep,
                            unsigned short* __restrict__ final_bf) {
    int seg = blockIdx.x;
    int s = sep[2 * seg], e = sep[2 * seg + 1];
    int len = e - s; if (len < 1) len = 1; if (len > 8) len = 8;
    float inv = 1.0f / (float)len;
    for (int c = threadIdx.x; c < kD; c += blockDim.x) {
        float acc = 0.f;
        const float* p = emb + (size_t)s * kD + c;
        for (int r = 0; r < len; ++r) acc += p[(size_t)r * kD];
        final_bf[(size_t)seg * kF + c] = f2bf(acc * inv);
    }
}

// ---------------------------------------------------------------
// Attention: one block (2 waves) per 16-row parity tile.
//   block b: parity = b&1, g = b>>1, rows = g*32 + 2*j + parity.
// Pass 1: scores = Q @ KV^T  (wave w handles key chunks 16w..16w+15)
// Softmax in LDS (fp32), probs repacked bf16.
// Pass 2: attended = P @ KV  (B-frags from KV^T copy for contiguity),
//   wave w handles d-chunks w, w+2, ...
// ---------------------------------------------------------------
__launch_bounds__(64)
__global__ void attn_kernel(unsigned short* __restrict__ final_bf,
                            const unsigned short* __restrict__ pp,
                            const unsigned short* __restrict__ yp,
                            const unsigned short* __restrict__ ppT,
                            const unsigned short* __restrict__ ypT) {
    __shared__ float          sc[16][520];   // fp32 scores, padded rows
    __shared__ unsigned short Pb[16][520];   // bf16 probabilities
    __shared__ float red[64], red2[64], rowm[16], rowsum[16];

    const int tid  = threadIdx.x;
    const int lane = tid & 31, wave = tid >> 5;
    const int l16  = lane & 15, h = lane >> 4;
    const int t = blockIdx.x;
    const int parity = t & 1, g = t >> 1;

    const unsigned short* kv  = parity ? yp  : pp;   // [512][768]
    const unsigned short* kvT = parity ? ypT : ppT;  // [768][512]

    // ---- Q A-fragments (held in registers; row = l16) ----
    const int qrow = g * 32 + 2 * l16 + parity;
    const unsigned short* qp = final_bf + (size_t)qrow * kF;
    AFrag a[24];
    #pragma unroll
    for (int kk = 0; kk < 24; ++kk) {
        const int d0 = kk * 32;
        a[kk].p.lo = *reinterpret_cast<const v8bf*>(qp + d0 + 16 * h);
        a[kk].p.hi = *reinterpret_cast<const v8bf*>(qp + d0 + 16 * h + 8);
    }

    // ---- QK^T ----
    for (int i = 0; i < 16; ++i) {
        const int n0 = (wave * 16 + i) * 16;
        const unsigned short* krow = kv + (size_t)(n0 + l16) * kD + 16 * h;
        v8f c = {};
        #pragma unroll
        for (int kk = 0; kk < 24; ++kk) {
            v16bf b = *reinterpret_cast<const v16bf*>(krow + kk * 32);
            c = __builtin_amdgcn_wmma_f32_16x16x32_bf16(false, a[kk].v,
                                                        false, b,
                                                        (short)0, c, false, false);
        }
        #pragma unroll
        for (int v = 0; v < 8; ++v) sc[v + 8 * h][n0 + l16] = c[v];
    }
    __syncthreads();

    // ---- softmax over 512 keys per row (4 threads per row) ----
    {
        const int row = tid & 15, q = tid >> 4;
        const int c0 = q * 128;
        float m = -3.4e38f;
        for (int c = 0; c < 128; ++c) m = fmaxf(m, sc[row][c0 + c]);
        red[tid] = m;
        __syncthreads();
        if (q == 0)
            rowm[row] = fmaxf(fmaxf(red[row], red[row + 16]),
                              fmaxf(red[row + 32], red[row + 48]));
        __syncthreads();
        const float mm = rowm[row];
        float s = 0.f;
        for (int c = 0; c < 128; ++c) s += __expf(sc[row][c0 + c] - mm);
        red2[tid] = s;
        __syncthreads();
        if (q == 0)
            rowsum[row] = red2[row] + red2[row + 16] + red2[row + 32] + red2[row + 48];
        __syncthreads();
        const float inv = 1.0f / rowsum[row];
        for (int c = 0; c < 128; ++c)
            Pb[row][c0 + c] = f2bf(__expf(sc[row][c0 + c] - mm) * inv);
    }
    __syncthreads();

    // ---- P @ KV  (attended) ----
    for (int dc = wave; dc < 48; dc += 2) {
        const int n0 = dc * 16;
        const unsigned short* vrow = kvT + (size_t)(n0 + l16) * kSP + 16 * h;
        v8f c = {};
        #pragma unroll
        for (int kc = 0; kc < 16; ++kc) {
            const int k0 = kc * 32;
            AFrag af;
            af.p.lo = *reinterpret_cast<const v8bf*>(&Pb[l16][k0 + 16 * h]);
            af.p.hi = *reinterpret_cast<const v8bf*>(&Pb[l16][k0 + 16 * h + 8]);
            v16bf b = *reinterpret_cast<const v16bf*>(vrow + k0);
            c = __builtin_amdgcn_wmma_f32_16x16x32_bf16(false, af.v,
                                                        false, b,
                                                        (short)0, c, false, false);
        }
        #pragma unroll
        for (int v = 0; v < 8; ++v) {
            const int M = v + 8 * h;
            const int r = g * 32 + 2 * M + parity;
            final_bf[(size_t)r * kF + kD + n0 + l16] = f2bf(c[v]);
        }
    }
}

// ---------------------------------------------------------------
// MLP: h = selu(final @ W1 + b1); out = softmax(h @ W2 + b2).
// One block (2 waves) per 16-row tile.  WMMA for final@W1 (K=1536),
// fp32 LDS h-tile, scalar 384x5 head.
// ---------------------------------------------------------------
__launch_bounds__(64)
__global__ void mlp_kernel(const unsigned short* __restrict__ final_bf,
                           const unsigned short* __restrict__ W1T,
                           const float* __restrict__ b1,
                           const float* __restrict__ W2,
                           const float* __restrict__ b2,
                           float* __restrict__ out) {
    __shared__ float hbuf[16][392];

    const int tid  = threadIdx.x;
    const int lane = tid & 31, wave = tid >> 5;
    const int l16  = lane & 15, h = lane >> 4;
    const int r0 = blockIdx.x * 16;

    const unsigned short* arow = final_bf + (size_t)(r0 + l16) * kF;

    for (int nc = wave; nc < 24; nc += 2) {
        const int n0 = nc * 16;
        const unsigned short* wrow = W1T + (size_t)(n0 + l16) * kF + 16 * h;
        v8f c = {};
        #pragma unroll
        for (int kc = 0; kc < 48; ++kc) {
            const int k0 = kc * 32;
            AFrag af;
            af.p.lo = *reinterpret_cast<const v8bf*>(arow + k0 + 16 * h);
            af.p.hi = *reinterpret_cast<const v8bf*>(arow + k0 + 16 * h + 8);
            v16bf b = *reinterpret_cast<const v16bf*>(wrow + k0);
            c = __builtin_amdgcn_wmma_f32_16x16x32_bf16(false, af.v,
                                                        false, b,
                                                        (short)0, c, false, false);
        }
        const float bias = b1[n0 + l16];
        constexpr float lam = 1.0507009873554805f;
        constexpr float alp = 1.6732632423543772f;
        #pragma unroll
        for (int v = 0; v < 8; ++v) {
            const float x = c[v] + bias;
            hbuf[v + 8 * h][n0 + l16] = (x > 0.f) ? lam * x
                                                  : lam * alp * (__expf(x) - 1.f);
        }
    }
    __syncthreads();

    if (tid < 16) {
        const int row = tid;
        float lg[kOut];
        #pragma unroll
        for (int o = 0; o < kOut; ++o) lg[o] = b2[o];
        for (int k = 0; k < kH1; ++k) {
            const float hv = hbuf[row][k];
            #pragma unroll
            for (int o = 0; o < kOut; ++o) lg[o] += hv * W2[k * kOut + o];
        }
        float m = lg[0];
        #pragma unroll
        for (int o = 1; o < kOut; ++o) m = fmaxf(m, lg[o]);
        float s = 0.f;
        #pragma unroll
        for (int o = 0; o < kOut; ++o) { lg[o] = __expf(lg[o] - m); s += lg[o]; }
        const float inv = 1.0f / s;
        #pragma unroll
        for (int o = 0; o < kOut; ++o)
            out[(size_t)(r0 + row) * kOut + o] = lg[o] * inv;
    }
}

// ---------------------------------------------------------------
extern "C" void kernel_launch(void* const* d_in, const int* in_sizes, int n_in,
                              void* d_out, int out_size, void* d_ws, size_t ws_size,
                              hipStream_t stream) {
    (void)in_sizes; (void)n_in; (void)out_size; (void)ws_size;
    const float* emb = (const float*)d_in[0];
    const float* pp  = (const float*)d_in[1];
    const float* yp  = (const float*)d_in[2];
    const float* W1  = (const float*)d_in[3];
    const float* b1  = (const float*)d_in[4];
    const float* W2  = (const float*)d_in[5];
    const float* b2  = (const float*)d_in[6];
    const int*   sep = (const int*)d_in[7];
    float* out = (float*)d_out;

    // workspace carve-up (~29.5 MB total)
    char* ws = (char*)d_ws;
    size_t off = 0;
    unsigned short* pp_bf  = (unsigned short*)(ws + off); off += (size_t)kSP * kD * 2;
    unsigned short* yp_bf  = (unsigned short*)(ws + off); off += (size_t)kSP * kD * 2;
    unsigned short* ppT_bf = (unsigned short*)(ws + off); off += (size_t)kSP * kD * 2;
    unsigned short* ypT_bf = (unsigned short*)(ws + off); off += (size_t)kSP * kD * 2;
    unsigned short* W1T_bf = (unsigned short*)(ws + off); off += (size_t)kF  * kH1 * 2;
    unsigned short* fin_bf = (unsigned short*)(ws + off); off += (size_t)kN  * kF  * 2;

    const int thr = 256;
    conv_kernel<<<(kSP * kD + thr - 1) / thr, thr, 0, stream>>>(pp, pp_bf, ppT_bf, kSP, kD);
    conv_kernel<<<(kSP * kD + thr - 1) / thr, thr, 0, stream>>>(yp, yp_bf, ypT_bf, kSP, kD);
    conv_kernel<<<(kF * kH1 + thr - 1) / thr, thr, 0, stream>>>(W1, nullptr, W1T_bf, kF, kH1);

    pool_kernel<<<kN, 256, 0, stream>>>(emb, sep, fin_bf);

    attn_kernel<<<kN / 16, 64, 0, stream>>>(fin_bf, pp_bf, yp_bf, ppT_bf, ypT_bf);

    mlp_kernel<<<kN / 16, 64, 0, stream>>>(fin_bf, W1T_bf, b1, W2, b2, out);
}
